// KQN_22179211116637
// MI455X (gfx1250) — matrix operand order repaired and checked by reading, not compiled, not run on hardware
//
#include <hip/hip_runtime.h>
#include <hip/hip_bf16.h>
#include <stdint.h>
#include <stddef.h>

// ---------------- problem constants ----------------
static constexpr int B_ = 32, T_ = 200, HID_ = 256, G_ = 1024, E_ = 128;
static constexpr int ITEM_DEPTH_ = 4001, IN_ = 4402;

// ---------------- workspace layout (bytes, 256B-aligned) ----------------
static constexpr size_t OFF_WT   = 0;                                   // W_ih^T fp32 [4402][1024]  18.0 MB
static constexpr size_t OFF_XP   = OFF_WT  + (size_t)IN_ * G_ * 4;      // xproj packed fp32          26.2 MB
static constexpr size_t OFF_WHH  = OFF_XP  + (size_t)T_ * 64 * 4 * 16 * 8 * 4; // W_hh bf16 packed    0.5 MB
static constexpr size_t OFF_LINW = OFF_WHH + (size_t)64 * 8 * 32 * 16 * 2;     // lin_W bf16 packed   64 KB
static constexpr size_t OFF_HS   = OFF_LINW+ (size_t)8 * 8 * 32 * 16 * 2;      // hs bf16 [T][32][256] 3.3 MB
static constexpr size_t OFF_Q    = OFF_HS  + (size_t)T_ * B_ * HID_ * 2;       // query fp32 [6400][128] 3.3 MB

// ---------------- WMMA types ----------------
typedef __attribute__((ext_vector_type(16))) __bf16 v16bf;
typedef __attribute__((ext_vector_type(8)))  float  v8f;
typedef __attribute__((ext_vector_type(4)))  int    v4i;   // POD 16B vector (works across addrspaces)
typedef __attribute__((ext_vector_type(4)))  float  v4f;
union AB2 { v4i q[2]; v16bf v; };

#define GLOBAL_AS __attribute__((address_space(1)))
typedef GLOBAL_AS const v4i g_v4i;

__device__ __forceinline__ unsigned short f2bf(float f) {
  unsigned int u = __float_as_uint(f);
  u += 0x7fffu + ((u >> 16) & 1u);           // round to nearest even
  return (unsigned short)(u >> 16);
}

// Branchless activations: the LSTM cell update is serialized between per-step
// barriers, so no EXEC save/restore or branches allowed on this path.
#if __has_builtin(__builtin_amdgcn_tanhf)
__device__ __forceinline__ float fast_tanh(float x) { return __builtin_amdgcn_tanhf(x); }
#else
__device__ __forceinline__ float fast_tanh(float x) {
  float e = __expf(2.0f * x);                // v_exp_f32
  return 1.0f - 2.0f * __builtin_amdgcn_rcpf(e + 1.0f); // v_rcp_f32
}
#endif
__device__ __forceinline__ float sigm(float x) {
  return __builtin_amdgcn_rcpf(1.0f + __expf(-x));
}

// ---------------- 1) transpose W_ih so one-hot gathers are contiguous rows ----------------
__global__ void k_transpose_wih(const float* __restrict__ w, float* __restrict__ wt) {
  __shared__ float tile[32][33];
  const int v0 = blockIdx.x * 32;            // input col (0..4401)
  const int g0 = blockIdx.y * 32;            // input row (0..1023)
  const int tx = threadIdx.x, ty = threadIdx.y;   // block (32,8)
  for (int i = ty; i < 32; i += 8) {
    int v = v0 + tx;
    tile[i][tx] = (v < IN_) ? w[(size_t)(g0 + i) * IN_ + v] : 0.f;
  }
  __syncthreads();
  for (int i = ty; i < 32; i += 8) {
    int v = v0 + i;
    if (v < IN_) wt[(size_t)v * G_ + (g0 + tx)] = tile[tx][i];
  }
}

// ---------------- 2) pack W_hh -> bf16 B-operand layout [nt64][ks8][lane32][e16] ----------------
// B(32x16) per-lane: lanes 0-15 hold K=0..7,16..23 (n=lane), lanes 16-31 hold K=8..15,24..31.
__global__ void k_pack_whh(const float* __restrict__ whh, unsigned short* __restrict__ out) {
  int idx = blockIdx.x * blockDim.x + threadIdx.x;       // 64*8*32*16 = 262144
  if (idx >= 64 * 8 * 32 * 16) return;
  int e = idx & 15, lane = (idx >> 4) & 31, ks = (idx >> 9) & 7, nt = idx >> 12;
  int n = nt * 16 + (lane & 15);
  int k = ks * 32 + (e & 7) + 16 * (e >> 3) + 8 * (lane >> 4);
  out[idx] = f2bf(whh[(size_t)n * HID_ + k]);
}

// ---------------- 3) pack lin_W -> bf16 B-operand layout [nt8][ks8][lane32][e16] ----------------
__global__ void k_pack_linw(const float* __restrict__ lw, unsigned short* __restrict__ out) {
  int idx = blockIdx.x * blockDim.x + threadIdx.x;       // 8*8*32*16 = 32768
  if (idx >= 8 * 8 * 32 * 16) return;
  int e = idx & 15, lane = (idx >> 4) & 31, ks = (idx >> 9) & 7, nt = idx >> 12;
  int n = nt * 16 + (lane & 15);                          // embed row 0..127
  int k = ks * 32 + (e & 7) + 16 * (e >> 3) + 8 * (lane >> 4);
  out[idx] = f2bf(lw[(size_t)n * HID_ + k]);
}

// ---------------- 4) materialize interleaved query [row=t*32+b][128] ----------------
__global__ void k_build_query(const int* __restrict__ item_ids, const int* __restrict__ skill_ids,
                              const float* __restrict__ item_emb, const float* __restrict__ skill_emb,
                              float* __restrict__ q) {
  int row = blockIdx.x;                                   // 0..6399
  int e = threadIdx.x;                                    // 0..127
  int t = row >> 5, b = row & 31;
  int u = e >> 1;
  float val = (e & 1) ? skill_emb[(size_t)skill_ids[b * T_ + t] * 64 + u]
                      : item_emb [(size_t)item_ids [b * T_ + t] * 64 + u];
  q[(size_t)row * E_ + e] = val;
}

// ---------------- 5) one-hot "GEMM" as a gather, pre-packed into WMMA C layout ----------------
// Packed: xp[t][nt64][bq4][l16][r8]; value at (b = 8*bq + r, n = 16*nt + l16).
__global__ void k_build_xproj(const int* __restrict__ item_in, const int* __restrict__ skill_in,
                              const float* __restrict__ wt,
                              const float* __restrict__ bih, const float* __restrict__ bhh,
                              float* __restrict__ xp) {
  int p = blockIdx.x;                                     // 0..6399 ; p = t*32 + b
  int t = p >> 5, b = p & 31;
  const float* wi  = wt + (size_t)item_in [b * T_ + t] * G_;
  const float* wsk = wt + (size_t)(ITEM_DEPTH_ + skill_in[b * T_ + t]) * G_;
  int bq = b >> 3, r = b & 7;
  for (int n = threadIdx.x; n < G_; n += blockDim.x) {
    float v = wi[n] + wsk[n] + bih[n] + bhh[n];
    int nt = n >> 4, l16 = n & 15;
    xp[((((size_t)t * 64 + nt) * 4 + bq) * 16 + l16) * 8 + r] = v;
  }
}

// ---------------- 6) persistent single-WGP LSTM: 32 waves, h in LDS, c in registers ----------
__global__ __launch_bounds__(1024) void k_lstm(
    const float* __restrict__ xp,             // packed input projection (+biases)
    const unsigned short* __restrict__ whh_p, // bf16 packed W_hh (L2-resident, streamed each step)
    unsigned short* __restrict__ hs,          // bf16 [T][32][256] output trajectory
    float* __restrict__ hT, float* __restrict__ cT) {
  __shared__ unsigned short hbuf[32 * 256];   // 16 KB current h (bf16, row-major [b][k])
  const int tid = threadIdx.x;
  const int wave = tid >> 5, lane = tid & 31;
  const int m_half = wave >> 4, n_base = wave & 15;
  const int hi = lane >> 4, l16 = lane & 15;
  const int jcol = 16 * n_base + l16;         // hidden unit handled by this lane
  const int bq = 2 * m_half + hi;             // row quad for packed xproj / C layout

  for (int i = tid; i < 32 * 256; i += 1024) hbuf[i] = 0;
  float c_st[8];                              // cell state for rows 8*bq + r, col jcol
  #pragma unroll
  for (int r = 0; r < 8; ++r) c_st[r] = 0.f;
  __syncthreads();

  // per-lane byte base inside hbuf for A-tile reads (two 16B chunks per k-step)
  const int a_base = 2 * ((16 * m_half + l16) * 256) + 16 * hi;

  for (int t = 0; t < T_; ++t) {
    // Opaque B base pointer per timestep (blocks LICM of 256 VGPRs of operands
    // and the scratch spills that caused), then addrspacecast back to global so
    // the loads lower to global_load_b128 (LOADcnt only) instead of flat_load
    // (LOADcnt + DScnt + LDS aperture path).
    const unsigned short* wp0 = whh_p;
    asm volatile("" : "+v"(wp0));
    GLOBAL_AS const unsigned short* wp = (GLOBAL_AS const unsigned short*)wp0;

    // accumulators init = x_proj (gathered one-hot projection + b_ih + b_hh)
    v8f acc[4];
    #pragma unroll
    for (int s = 0; s < 4; ++s) {
      int nt = 16 * s + n_base;
      const float* p = xp + ((((size_t)t * 64 + nt) * 4 + bq) * 16 + l16) * 8;
      v4f x0 = *(const v4f*)p;
      v4f x1 = *(const v4f*)(p + 4);
      acc[s] = (v8f){x0.x, x0.y, x0.z, x0.w, x1.x, x1.y, x1.z, x1.w};
    }

    // gates += h @ W_hh^T via bf16 WMMA (K = 256 -> 8 k-steps of 32)
    #pragma unroll 2
    for (int ks = 0; ks < 8; ++ks) {
      // A tile: rows 16*m_half..+15, K = 32*ks..+31 (two ds_load_b128)
      AB2 a;
      const char* ap = (const char*)hbuf + a_base + 64 * ks;
      a.q[0] = *(const v4i*)(ap);
      a.q[1] = *(const v4i*)(ap + 32);
      // issue all four B tiles first so loads clause together, then 4 WMMAs
      AB2 b0, b1, b2, b3;
      {
        const g_v4i* p0 = (const g_v4i*)(wp + ((((size_t)(16 * 0 + n_base)) * 8 + ks) * 32 + lane) * 16);
        const g_v4i* p1 = (const g_v4i*)(wp + ((((size_t)(16 * 1 + n_base)) * 8 + ks) * 32 + lane) * 16);
        const g_v4i* p2 = (const g_v4i*)(wp + ((((size_t)(16 * 2 + n_base)) * 8 + ks) * 32 + lane) * 16);
        const g_v4i* p3 = (const g_v4i*)(wp + ((((size_t)(16 * 3 + n_base)) * 8 + ks) * 32 + lane) * 16);
        b0.q[0] = p0[0]; b0.q[1] = p0[1];
        b1.q[0] = p1[0]; b1.q[1] = p1[1];
        b2.q[0] = p2[0]; b2.q[1] = p2[1];
        b3.q[0] = p3[0]; b3.q[1] = p3[1];
      }
      acc[0] = __builtin_amdgcn_wmma_f32_16x16x32_bf16(false, a.v, false, b0.v, (short)0, acc[0], false, false);
      acc[1] = __builtin_amdgcn_wmma_f32_16x16x32_bf16(false, a.v, false, b1.v, (short)0, acc[1], false, false);
      acc[2] = __builtin_amdgcn_wmma_f32_16x16x32_bf16(false, a.v, false, b2.v, (short)0, acc[2], false, false);
      acc[3] = __builtin_amdgcn_wmma_f32_16x16x32_bf16(false, a.v, false, b3.v, (short)0, acc[3], false, false);
    }
    __syncthreads();                          // everyone done reading old h
    // LSTM cell update, entirely in registers (acc[0..3] = i,f,g,o at same (row,jcol)).
    // All-branchless: hardware v_tanh / v_exp / v_rcp, no EXEC manipulation.
    #pragma unroll
    for (int r = 0; r < 8; ++r) {
      int row = 8 * bq + r;                   // batch index
      float ig = sigm(acc[0][r]);
      float fg = sigm(acc[1][r]);
      float gg = fast_tanh(acc[2][r]);
      float og = sigm(acc[3][r]);
      float c  = fg * c_st[r] + ig * gg;
      c_st[r]  = c;
      float h  = og * fast_tanh(c);
      unsigned short hb = f2bf(h);
      hbuf[row * 256 + jcol] = hb;
      hs[((size_t)t * 32 + row) * 256 + jcol] = hb;
      if (t == T_ - 1) { hT[row * 256 + jcol] = h; cT[row * 256 + jcol] = c; }
    }
    __syncthreads();                          // new h visible before next step
  }
}

// ---------------- 7) head: knowledge = hs @ lin_W^T + lin_b ; out = <knowledge, query> ----
__global__ __launch_bounds__(32) void k_final(
    const unsigned short* __restrict__ hs, const unsigned short* __restrict__ linw_p,
    const float* __restrict__ lin_b, const float* __restrict__ q, float* __restrict__ out) {
  __shared__ float red[16][17];
  const int lane = threadIdx.x;
  const int hi = lane >> 4, l16 = lane & 15;
  const int rowbase = blockIdx.x * 16;        // 16 rows of [t*32+b]
  v8f acc[8];
  #pragma unroll
  for (int s = 0; s < 8; ++s) acc[s] = (v8f){0.f, 0.f, 0.f, 0.f, 0.f, 0.f, 0.f, 0.f};
  const char* ab = (const char*)(hs + (size_t)(rowbase + l16) * HID_) + 16 * hi;
  for (int ks = 0; ks < 8; ++ks) {
    AB2 a;
    a.q[0] = *(const v4i*)(ab + 64 * ks);
    a.q[1] = *(const v4i*)(ab + 64 * ks + 32);
    #pragma unroll
    for (int s = 0; s < 8; ++s) {
      const v4i* bp = (const v4i*)(linw_p + (((size_t)s * 8 + ks) * 32 + lane) * 16);
      AB2 bm;
      bm.q[0] = bp[0]; bm.q[1] = bp[1];
      acc[s] = __builtin_amdgcn_wmma_f32_16x16x32_bf16(
          false, a.v, false, bm.v, (short)0, acc[s], false, false);
    }
  }
  float part[8];
  #pragma unroll
  for (int r = 0; r < 8; ++r) part[r] = 0.f;
  #pragma unroll
  for (int s = 0; s < 8; ++s) {
    int col = 16 * s + l16;
    float lb = lin_b[col];
    #pragma unroll
    for (int r = 0; r < 8; ++r) {
      int row = rowbase + r + 8 * hi;
      part[r] += (acc[s][r] + lb) * q[(size_t)row * E_ + col];
    }
  }
  #pragma unroll
  for (int r = 0; r < 8; ++r) red[r + 8 * hi][l16] = part[r];
  __syncthreads();
  if (lane < 16) {
    float sacc = 0.f;
    #pragma unroll
    for (int j = 0; j < 16; ++j) sacc += red[lane][j];
    int row = rowbase + lane;
    int t = row >> 5, b = row & 31;
    out[b * T_ + t] = sacc;                   // reference output layout [B,T]
  }
}

// ---------------- launch ----------------
extern "C" void kernel_launch(void* const* d_in, const int* in_sizes, int n_in,
                              void* d_out, int out_size, void* d_ws, size_t ws_size,
                              hipStream_t stream) {
  const int*   item_in   = (const int*)  d_in[0];
  const int*   skill_in  = (const int*)  d_in[1];
  const int*   item_ids  = (const int*)  d_in[2];
  const int*   skill_ids = (const int*)  d_in[3];
  const float* W_ih      = (const float*)d_in[4];
  const float* W_hh      = (const float*)d_in[5];
  const float* b_ih      = (const float*)d_in[6];
  const float* b_hh      = (const float*)d_in[7];
  const float* lin_W     = (const float*)d_in[8];
  const float* lin_b     = (const float*)d_in[9];
  const float* item_emb  = (const float*)d_in[10];
  const float* skill_emb = (const float*)d_in[11];

  char* ws = (char*)d_ws;
  float*          wT     = (float*)(ws + OFF_WT);
  float*          xp     = (float*)(ws + OFF_XP);
  unsigned short* whh_p  = (unsigned short*)(ws + OFF_WHH);
  unsigned short* linw_p = (unsigned short*)(ws + OFF_LINW);
  unsigned short* hs     = (unsigned short*)(ws + OFF_HS);
  float*          q      = (float*)(ws + OFF_Q);

  float* out = (float*)d_out;                 // [B*T] output
  float* hT  = out + B_ * T_;                 // [1,B,H]
  float* cT  = hT + B_ * HID_;                // [1,B,H]

  k_transpose_wih<<<dim3((IN_ + 31) / 32, G_ / 32), dim3(32, 8), 0, stream>>>(W_ih, wT);
  k_pack_whh  <<<(64 * 8 * 32 * 16) / 256, 256, 0, stream>>>(W_hh, whh_p);
  k_pack_linw <<<(8 * 8 * 32 * 16) / 256, 256, 0, stream>>>(lin_W, linw_p);
  k_build_query<<<B_ * T_, 128, 0, stream>>>(item_ids, skill_ids, item_emb, skill_emb, q);
  k_build_xproj<<<B_ * T_, 256, 0, stream>>>(item_in, skill_in, wT, b_ih, b_hh, xp);
  k_lstm<<<1, 1024, 0, stream>>>(xp, whh_p, hs, hT, cT);
  k_final<<<(B_ * T_) / 16, 32, 0, stream>>>(hs, linw_p, lin_b, q, out);
}
